// Trm_Encoder_83365315215702
// MI455X (gfx1250) — matrix operand order repaired and checked by reading, not compiled, not run on hardware
//
#include <hip/hip_runtime.h>
#include <hip/hip_bf16.h>

typedef _Float16 f16;
typedef _Float16 v8h  __attribute__((ext_vector_type(8)));
typedef _Float16 v16h __attribute__((ext_vector_type(16)));
typedef float    v8f  __attribute__((ext_vector_type(8)));
typedef unsigned int v4u __attribute__((ext_vector_type(4)));
typedef int v4i __attribute__((ext_vector_type(4)));
typedef int v8i __attribute__((ext_vector_type(8)));

#define BB 4
#define LL 1024
#define DD 300
#define HH 6
#define DPH 50
#define NLAYER 6
#define MROWS (BB*LL)   /* 4096 */
#define BHD   (BB*HH)   /* 24 batch-heads */
#define SCALE_QK 0.35355339059327373f   /* (50//6)^-0.5 = 8^-0.5, faithful quirk */

__device__ __forceinline__ float act_apply(float v, int act) {
  if (act == 1) return v > 0.0f ? v : 0.0f;
  if (act == 2) {
    const float al = 1.6732632423543772f, sc = 1.0507009873554805f;
    return v > 0.0f ? sc * v : sc * al * (__expf(v) - 1.0f);
  }
  return v;
}

// ---------------------------------------------------------------------------
// TDM: DMA a 2D f16 tile (rows x cols elements, row stride == cols) from
// global memory into LDS. Descriptor packed per CDNA5 ISA 08_async_tensor
// sections 8.3/8.4: group0 {count=1, lds_addr, global_addr, type=2},
// group1 {data_size=2B, tensor dims == tile dims, dim0 stride}, groups 2/3
// zero (2D tensor, no iterate/gather/multicast).
// ---------------------------------------------------------------------------
#if __has_builtin(__builtin_amdgcn_tensor_load_to_lds)
#define HAVE_TDM 1
__device__ __forceinline__ void tdm_load_tile(const f16* gsrc, unsigned lds_off,
                                              unsigned rows, unsigned cols) {
  unsigned long long ga = (unsigned long long)(size_t)gsrc;
  v4u g0;
  g0[0] = 1u;                                            // count=1 (valid), user mode
  g0[1] = lds_off;                                       // lds_addr (bytes)
  g0[2] = (unsigned)(ga & 0xFFFFFFFFu);                  // global_addr[31:0]
  g0[3] = (unsigned)((ga >> 32) & 0x01FFFFFFu) | 0x80000000u; // addr[56:32] | type=2
  v8i g1;
  g1[0] = 0x00010000;                                    // data_size=1 -> 2 bytes
  g1[1] = (int)((cols & 0xFFFFu) << 16);                 // tensor_dim0[15:0]
  g1[2] = (int)(((cols >> 16) & 0xFFFFu) | ((rows & 0xFFFFu) << 16)); // td0 hi | td1 lo
  g1[3] = (int)(((rows >> 16) & 0xFFFFu) | ((cols & 0xFFFFu) << 16)); // td1 hi | tile_dim0
  g1[4] = (int)(rows & 0xFFFFu);                         // tile_dim1 | tile_dim2=0
  g1[5] = (int)cols;                                     // tensor_dim0_stride[31:0]
  g1[6] = 0;                                             // stride hi | dim1_stride lo
  g1[7] = 0;
  v4i z4 = {0, 0, 0, 0};
#if defined(__clang_major__) && __clang_major__ >= 23
  v8i z8 = {0, 0, 0, 0, 0, 0, 0, 0};
  __builtin_amdgcn_tensor_load_to_lds(g0, g1, z4, z4, z8, 0);
#else
  __builtin_amdgcn_tensor_load_to_lds(g0, g1, z4, z4, 0);
#endif
}
#endif

// ---------------------------------------------------------------------------
// Weight convert: f32 [K,N] row-major  ->  f16 [Npad, Kpad] (pre-transposed,
// zero padded) so WMMA B-fragments are contiguous-K b128 loads.
// ---------------------------------------------------------------------------
__global__ void k_convw(const float* __restrict__ W, f16* __restrict__ T,
                        int K, int N, int Kpad, int Npad) {
  int t = blockIdx.x * blockDim.x + threadIdx.x;
  int total = Npad * Kpad;
  if (t >= total) return;
  int n = t / Kpad, k = t - n * Kpad;
  float v = (n < N && k < K) ? W[(size_t)k * N + n] : 0.0f;
  T[t] = (f16)v;
}

// ---------------------------------------------------------------------------
// Embedding gather + sinusoidal positional encoding.
// ---------------------------------------------------------------------------
__global__ void k_embed(const int* __restrict__ tok, const float* __restrict__ emb,
                        float* __restrict__ x, f16* __restrict__ xh) {
  int t = blockIdx.x * blockDim.x + threadIdx.x;
  if (t >= MROWS * 320) return;
  int row = t / 320, col = t - row * 320;
  if (col >= DD) { xh[t] = (f16)0.0f; return; }
  int l = row & (LL - 1);
  float expo = (2.0f * (float)(col >> 1)) / (float)DD;
  float ang = (float)l * __powf(10000.0f, -expo);
  float pe = (col & 1) ? __cosf(ang) : __sinf(ang);
  float v = emb[(size_t)tok[row] * DD + col] + pe;
  x[(size_t)row * DD + col] = v;
  xh[t] = (f16)v;
}

// ---------------------------------------------------------------------------
// Generic WMMA GEMM:  C[4096,N] = act(A[4096,Kpad](f16) @ Wt^T + bias)
// Wt pre-transposed [Npad,Kpad] f16.  One wave = one 16x16 tile, 8 waves/WG.
// The block's shared 16-row A tile is DMA'd once into LDS by the Tensor Data
// Mover (TENSORcnt), then A-fragments are ds_loads; B streams from global.
// ---------------------------------------------------------------------------
__global__ void k_gemm_wmma(const f16* __restrict__ A, const f16* __restrict__ Wt,
                            const float* __restrict__ bias,
                            float* __restrict__ Cf, int N,
                            f16* __restrict__ Ch, int ldh,
                            int ntiles, int Kpad, int act) {
  extern __shared__ char smemg[];
  f16* sA = (f16*)smemg;                       // 16 x Kpad f16 tile
  int wave = threadIdx.x >> 5;
  int lane = threadIdx.x & 31;
  int nt = blockIdx.y * 8 + wave;
  int m0 = blockIdx.x * 16;

  // ---- stage A tile (shared by all 8 waves) into LDS ----
#ifdef HAVE_TDM
  if (threadIdx.x == 0) {
    tdm_load_tile(A + (size_t)m0 * Kpad, (unsigned)(size_t)(void*)sA, 16u, (unsigned)Kpad);
  }
#if __has_builtin(__builtin_amdgcn_s_wait_tensorcnt)
  __builtin_amdgcn_s_wait_tensorcnt(0);
#else
  asm volatile("s_wait_tensorcnt 0" ::: "memory");
#endif
#else
  {
    const v8h* gsrc = (const v8h*)(A + (size_t)m0 * Kpad);
    v8h* dst = (v8h*)sA;
    int nvec = 16 * Kpad / 8;
    for (int i = threadIdx.x; i < nvec; i += blockDim.x) dst[i] = gsrc[i];
  }
#endif
  __syncthreads();
  if (nt >= ntiles) return;

  int n0 = nt * 16;
  int hi = (lane & 16) ? 1 : 0;
  int l15 = lane & 15;
  int ncol = n0 + l15;
  // A fragment from LDS: lanes 0-15 row M=lane, K=k0..k0+15;
  //                      lanes 16-31 row M=lane-16, K=k0+16..k0+31.
  const f16* pa = sA + (size_t)l15 * Kpad + (hi ? 16 : 0);
  // B fragment: lanes 0-15 col N=lane K=k0..k0+15; lanes 16-31 K=k0+16..k0+31.
  const f16* pb = Wt + (size_t)ncol * Kpad + (hi ? 16 : 0);
  v8f c = {};
  for (int k0 = 0; k0 < Kpad; k0 += 32) {
    __builtin_prefetch(pb + k0 + 128, 0, 1);
    v8h a0 = *(const v8h*)(pa + k0);
    v8h a1 = *(const v8h*)(pa + k0 + 8);
    v16h av, bv;
    bv = *(const v16h*)(pb + k0);
#pragma unroll
    for (int i = 0; i < 8; ++i) { av[i] = a0[i]; av[8 + i] = a1[i]; }
    c = __builtin_amdgcn_wmma_f32_16x16x32_f16(false, av, false, bv,
                                               (short)0, c, false, false);
  }
  float bvn = (ncol < N) ? bias[ncol] : 0.0f;
  int mbase = m0 + (hi ? 8 : 0);
#pragma unroll
  for (int r = 0; r < 8; ++r) {
    float v = act_apply(c[r] + bvn, act);
    int m = mbase + r;
    if (ncol < N && Cf) Cf[(size_t)m * N + ncol] = v;
    if (Ch) Ch[(size_t)m * ldh + ncol] = (ncol < N) ? (f16)v : (f16)0.0f;
  }
}

// ---------------------------------------------------------------------------
// Pack q/k/v f32 (quirky flat [24,1024,50] view of [4096,300]) into
// q_h/k_h [24,1024,64] f16 (d-padded) and v transposed [24,64,1024] f16.
// ---------------------------------------------------------------------------
__global__ void k_pack_qkv(const float* __restrict__ qf, const float* __restrict__ kf,
                           const float* __restrict__ vf,
                           f16* __restrict__ qh, f16* __restrict__ kh,
                           f16* __restrict__ vt) {
  int t = blockIdx.x * blockDim.x + threadIdx.x;
  if (t >= BHD * LL * 64) return;
  int d = t & 63;
  int row = (t >> 6) & (LL - 1);
  int bh = t >> 16;
  float qv = 0.0f, kv = 0.0f, vv = 0.0f;
  if (d < DPH) {
    int src = bh * (LL * DPH) + row * DPH + d;
    qv = qf[src]; kv = kf[src]; vv = vf[src];
  }
  qh[t] = (f16)qv;
  kh[t] = (f16)kv;
  vt[((size_t)bh * 64 + d) * LL + row] = (f16)vv;
}

// ---------------------------------------------------------------------------
// Attention: one wave32 per (batch-head, 16-row q tile). Scores (16x1024 f32)
// and probs (16x1024 f16) live entirely in LDS (96KB of the 320KB WGP LDS);
// the 100MB score tensor never touches HBM. QK^T and P@V both via WMMA.
// Mask batch index is bh % B (faithful to jnp.tile quirk).
// ---------------------------------------------------------------------------
__global__ void k_attn(const f16* __restrict__ qh, const f16* __restrict__ kh,
                       const f16* __restrict__ vt, const int* __restrict__ tok,
                       f16* __restrict__ ctxh) {
  extern __shared__ char smem[];
  float* sc = (float*)smem;                       // 16*1024 f32 = 64KB
  f16*   ph = (f16*)(smem + 16 * LL * 4);         // 16*1024 f16 = 32KB
  int bh = blockIdx.x >> 6;
  int m0 = (blockIdx.x & 63) * 16;
  int lane = threadIdx.x;
  int hi = (lane & 16) ? 1 : 0;
  int l15 = lane & 15;

  const f16* qbase = qh + ((size_t)bh * LL + (m0 + l15)) * 64 + (hi ? 16 : 0);
  v16h aq0, aq1;
  {
    v8h t0 = *(const v8h*)(qbase + 0);
    v8h t1 = *(const v8h*)(qbase + 8);
    v8h t2 = *(const v8h*)(qbase + 32);
    v8h t3 = *(const v8h*)(qbase + 40);
#pragma unroll
    for (int i = 0; i < 8; ++i) {
      aq0[i] = t0[i]; aq0[8 + i] = t1[i];
      aq1[i] = t2[i]; aq1[8 + i] = t3[i];
    }
  }

  for (int kt = 0; kt < 64; ++kt) {
    const f16* kb = kh + ((size_t)bh * LL + (kt * 16 + l15)) * 64 + (hi ? 16 : 0);
    v16h b0 = *(const v16h*)(kb + 0);
    v16h b1 = *(const v16h*)(kb + 32);
    v8f c = {};
    c = __builtin_amdgcn_wmma_f32_16x16x32_f16(false, aq0, false, b0, (short)0, c, false, false);
    c = __builtin_amdgcn_wmma_f32_16x16x32_f16(false, aq1, false, b1, (short)0, c, false, false);
    int ki = kt * 16 + l15;
    bool pd = (tok[(bh & 3) * LL + ki] == 0);
#pragma unroll
    for (int r = 0; r < 8; ++r) {
      float s = pd ? -1e9f : c[r] * SCALE_QK;
      sc[(r + (hi ? 8 : 0)) * LL + ki] = s;
    }
  }
  __syncthreads();

  for (int r = 0; r < 16; ++r) {
    float* row = sc + r * LL;
    float mx = -3.0e38f;
    for (int cix = lane; cix < LL; cix += 32) mx = fmaxf(mx, row[cix]);
#pragma unroll
    for (int o = 16; o > 0; o >>= 1) mx = fmaxf(mx, __shfl_xor(mx, o, 32));
    float sum = 0.0f;
    for (int cix = lane; cix < LL; cix += 32) {
      float e = __expf(row[cix] - mx); row[cix] = e; sum += e;
    }
#pragma unroll
    for (int o = 16; o > 0; o >>= 1) sum += __shfl_xor(sum, o, 32);
    float inv = 1.0f / sum;
    for (int cix = lane; cix < LL; cix += 32) ph[r * LL + cix] = (f16)(row[cix] * inv);
  }
  __syncthreads();

  v8f acc0 = {}, acc1 = {}, acc2 = {}, acc3 = {};
  const f16* prow = ph + l15 * LL + (hi ? 16 : 0);
  const f16* vbase = vt + ((size_t)bh * 64 + l15) * LL + (hi ? 16 : 0);
  for (int kk0 = 0; kk0 < LL; kk0 += 32) {
    v8h t0 = *(const v8h*)(prow + kk0);
    v8h t1 = *(const v8h*)(prow + kk0 + 8);
    v16h av;
#pragma unroll
    for (int i = 0; i < 8; ++i) { av[i] = t0[i]; av[8 + i] = t1[i]; }
    v16h b0 = *(const v16h*)(vbase + kk0);
    v16h b1 = *(const v16h*)(vbase + kk0 + 16 * LL);
    v16h b2 = *(const v16h*)(vbase + kk0 + 32 * LL);
    v16h b3 = *(const v16h*)(vbase + kk0 + 48 * LL);
    acc0 = __builtin_amdgcn_wmma_f32_16x16x32_f16(false, av, false, b0, (short)0, acc0, false, false);
    acc1 = __builtin_amdgcn_wmma_f32_16x16x32_f16(false, av, false, b1, (short)0, acc1, false, false);
    acc2 = __builtin_amdgcn_wmma_f32_16x16x32_f16(false, av, false, b2, (short)0, acc2, false, false);
    acc3 = __builtin_amdgcn_wmma_f32_16x16x32_f16(false, av, false, b3, (short)0, acc3, false, false);
  }
  auto wb = [&](const v8f& a, int ntl) {
    int dcol = ntl * 16 + l15;
    if (dcol >= DPH) return;
#pragma unroll
    for (int r = 0; r < 8; ++r) {
      int m = m0 + r + (hi ? 8 : 0);
      int flat = bh * (LL * DPH) + m * DPH + dcol;   // inverse reshape is flat identity
      int r2 = flat / DD, c2 = flat - r2 * DD;
      ctxh[(size_t)r2 * 320 + c2] = (f16)a[r];
    }
  };
  wb(acc0, 0); wb(acc1, 1); wb(acc2, 2); wb(acc3, 3);
}

// ---------------------------------------------------------------------------
// x = LayerNorm(x + delta)*g + b ; also emit f16 K-padded copy.
// ---------------------------------------------------------------------------
__global__ void k_add_ln(float* __restrict__ x, const float* __restrict__ dlt,
                         const float* __restrict__ g, const float* __restrict__ b,
                         f16* __restrict__ xh) {
  __shared__ float ybuf[DD];
  __shared__ float red[128];
  int row = blockIdx.x, tid = threadIdx.x;
  float s = 0.0f;
  for (int c = tid; c < DD; c += 128) {
    float v = x[(size_t)row * DD + c] + dlt[(size_t)row * DD + c];
    ybuf[c] = v; s += v;
  }
  red[tid] = s; __syncthreads();
  for (int o = 64; o > 0; o >>= 1) { if (tid < o) red[tid] += red[tid + o]; __syncthreads(); }
  float mean = red[0] / (float)DD; __syncthreads();
  float vs = 0.0f;
  for (int c = tid; c < DD; c += 128) { float dv = ybuf[c] - mean; vs += dv * dv; }
  red[tid] = vs; __syncthreads();
  for (int o = 64; o > 0; o >>= 1) { if (tid < o) red[tid] += red[tid + o]; __syncthreads(); }
  float inv = rsqrtf(red[0] / (float)DD + 1e-5f); __syncthreads();
  for (int c = tid; c < DD; c += 128) {
    float v = (ybuf[c] - mean) * inv * g[c] + b[c];
    x[(size_t)row * DD + c] = v;
    xh[(size_t)row * 320 + c] = (f16)v;
  }
  for (int c = DD + tid; c < 320; c += 128) xh[(size_t)row * 320 + c] = (f16)0.0f;
}

// out[2,B,D]: out[j,b,d] = sum_l X[b,l,d]*W[l,j] + bias[j]
__global__ void k_reduce_lw(const float* __restrict__ X, const float* __restrict__ W,
                            const float* __restrict__ bias, float* __restrict__ out) {
  int t = blockIdx.x * blockDim.x + threadIdx.x;
  if (t >= 2 * BB * DD) return;
  int j = t / (BB * DD);
  int rem = t - j * BB * DD;
  int b = rem / DD, d = rem - b * DD;
  float s = 0.0f;
  for (int l = 0; l < LL; ++l) s += X[((size_t)b * LL + l) * DD + d] * W[l * 2 + j];
  out[t] = s + bias[j];
}

// ---------------------------------------------------------------------------
extern "C" void kernel_launch(void* const* d_in, const int* in_sizes, int n_in,
                              void* d_out, int out_size, void* d_ws, size_t ws_size,
                              hipStream_t stream) {
  (void)in_sizes; (void)n_in; (void)out_size; (void)ws_size;
  const int*   tok   = (const int*)  d_in[0];
  const float* emb   = (const float*)d_in[2];
  const float* Wq    = (const float*)d_in[3];
  const float* bq    = (const float*)d_in[4];
  const float* Wk    = (const float*)d_in[5];
  const float* bk    = (const float*)d_in[6];
  const float* Wv    = (const float*)d_in[7];
  const float* bv    = (const float*)d_in[8];
  const float* Wo    = (const float*)d_in[9];
  const float* bo    = (const float*)d_in[10];
  const float* ln1g  = (const float*)d_in[11];
  const float* ln1b  = (const float*)d_in[12];
  const float* W1    = (const float*)d_in[13];
  const float* b1    = (const float*)d_in[14];
  const float* W2    = (const float*)d_in[15];
  const float* b2    = (const float*)d_in[16];
  const float* ln2g  = (const float*)d_in[17];
  const float* ln2b  = (const float*)d_in[18];
  const float* enc1W = (const float*)d_in[19];
  const float* enc1b = (const float*)d_in[20];
  const float* enc2W = (const float*)d_in[21];
  const float* enc2b = (const float*)d_in[22];
  const float* WOw   = (const float*)d_in[23];
  const float* WOb   = (const float*)d_in[24];
  const float* Wcc   = (const float*)d_in[25];
  const float* bcc   = (const float*)d_in[26];
  const float* Wcc2  = (const float*)d_in[27];
  const float* bcc2  = (const float*)d_in[28];
  const float* out1W = (const float*)d_in[29];
  const float* out1b = (const float*)d_in[30];
  const float* out2W = (const float*)d_in[31];
  const float* out2b = (const float*)d_in[32];
  const float* WAw   = (const float*)d_in[33];
  const float* WAb   = (const float*)d_in[34];
  const float* Wm    = (const float*)d_in[35];
  const float* bm    = (const float*)d_in[36];
  const float* Wn    = (const float*)d_in[37];
  const float* bn    = (const float*)d_in[38];

  char* wsb = (char*)d_ws;
  size_t off = 0;
  auto alloc = [&](size_t bytes) -> void* {
    void* p = wsb + off;
    off += (bytes + 255) & ~(size_t)255;
    return p;
  };
  auto convw = [&](const float* W, int K, int N) -> f16* {
    int Kpad = (K + 31) & ~31, Npad = (N + 15) & ~15;
    f16* T = (f16*)alloc((size_t)Kpad * Npad * sizeof(f16));
    int total = Kpad * Npad;
    k_convw<<<(total + 255) / 256, 256, 0, stream>>>(W, T, K, N, Kpad, Npad);
    return T;
  };

  f16 *WqT[NLAYER], *WkT[NLAYER], *WvT[NLAYER], *WoT[NLAYER], *W1T[NLAYER], *W2T[NLAYER];
  for (int i = 0; i < NLAYER; ++i) {
    WqT[i] = convw(Wq + (size_t)i * DD * DD, DD, DD);
    WkT[i] = convw(Wk + (size_t)i * DD * DD, DD, DD);
    WvT[i] = convw(Wv + (size_t)i * DD * DD, DD, DD);
    WoT[i] = convw(Wo + (size_t)i * DD * DD, DD, DD);
    W1T[i] = convw(W1 + (size_t)i * DD * 1200, DD, 1200);
    W2T[i] = convw(W2 + (size_t)i * 1200 * DD, 1200, DD);
  }
  f16* enc1T = convw(enc1W, DD, DD);
  f16* enc2T = convw(enc2W, DD, DD);
  f16* WOT   = convw(WOw, DD, 1200);
  f16* WccT  = convw(Wcc, DD, 600);
  f16* Wcc2T = convw(Wcc2, 600, DD);
  f16* out1T = convw(out1W, 1500, 1500);
  f16* out2T = convw(out2W, 1500, 1200);
  f16* WAT   = convw(WAw, 1200, 1200);

  float* x    = (float*)alloc((size_t)MROWS * DD * 4);
  f16*   xh   = (f16*)  alloc((size_t)MROWS * 320 * 2);
  float* qf   = (float*)alloc((size_t)MROWS * DD * 4);
  float* kf   = (float*)alloc((size_t)MROWS * DD * 4);
  float* vf   = (float*)alloc((size_t)MROWS * DD * 4);
  f16*   qhB  = (f16*)  alloc((size_t)BHD * LL * 64 * 2);
  f16*   khB  = (f16*)  alloc((size_t)BHD * LL * 64 * 2);
  f16*   vtB  = (f16*)  alloc((size_t)BHD * 64 * LL * 2);
  f16*   ctxh = (f16*)  alloc((size_t)MROWS * 320 * 2);
  f16*   f1h  = (f16*)  alloc((size_t)MROWS * 1200 * 2);
  f16*   g1h  = (f16*)  alloc((size_t)MROWS * 320 * 2);
  f16*   cath = (f16*)  alloc((size_t)MROWS * 1504 * 2);
  f16*   oh1  = (f16*)  alloc((size_t)MROWS * 1504 * 2);
  f16*   oh2  = (f16*)  alloc((size_t)MROWS * 1200 * 2);
  f16*   cc1h = (f16*)  alloc((size_t)MROWS * 608 * 2);

  (void)hipMemsetAsync(ctxh, 0, (size_t)MROWS * 320 * 2, stream);
  (void)hipMemsetAsync(g1h,  0, (size_t)MROWS * 320 * 2, stream);
  (void)hipMemsetAsync(cath, 0, (size_t)MROWS * 1504 * 2, stream);

  auto gemm = [&](const f16* A, const f16* Wt, const float* bias, float* Cf, int N,
                  f16* Ch, int ldh, int Kpad, int act) {
    int ntiles = (N + 15) / 16;
    dim3 grid(MROWS / 16, (ntiles + 7) / 8);
    size_t shmem = (size_t)16 * Kpad * sizeof(f16);
    k_gemm_wmma<<<grid, 256, shmem, stream>>>(A, Wt, bias, Cf, N, Ch, ldh, ntiles, Kpad, act);
  };

  k_embed<<<(MROWS * 320 + 255) / 256, 256, 0, stream>>>(tok, emb, x, xh);

  for (int i = 0; i < NLAYER; ++i) {
    gemm(xh, WqT[i], bq + i * DD, qf, DD, nullptr, 0, 320, 0);
    gemm(xh, WkT[i], bk + i * DD, kf, DD, nullptr, 0, 320, 0);
    gemm(xh, WvT[i], bv + i * DD, vf, DD, nullptr, 0, 320, 0);
    k_pack_qkv<<<(BHD * LL * 64 + 255) / 256, 256, 0, stream>>>(qf, kf, vf, qhB, khB, vtB);
    k_attn<<<BHD * 64, 32, 96 * 1024, stream>>>(qhB, khB, vtB, tok, ctxh);
    gemm(ctxh, WoT[i], bo + i * DD, qf, DD, nullptr, 0, 320, 0);
    k_add_ln<<<MROWS, 128, 0, stream>>>(x, qf, ln1g + i * DD, ln1b + i * DD, xh);
    gemm(xh, W1T[i], b1 + i * 1200, nullptr, 1200, f1h, 1200, 320, 1);   // relu
    gemm(f1h, W2T[i], b2 + i * DD, kf, DD, nullptr, 0, 1200, 0);
    k_add_ln<<<MROWS, 128, 0, stream>>>(x, kf, ln2g + i * DD, ln2b + i * DD, xh);
  }

  float* out_o    = (float*)d_out;
  float* out_feat = out_o + (size_t)MROWS * 1200;
  float* out_h    = out_feat + (size_t)MROWS * 1200;
  float* out_c    = out_h + 2 * BB * DD;

  gemm(xh, enc1T, enc1b, nullptr, DD, g1h, 320, 320, 2);
  gemm(g1h, enc2T, enc2b, nullptr, DD, cath, 1504, 320, 2);
  gemm(xh, WOT, WOb, nullptr, 1200, cath + 300, 1504, 320, 0);
  gemm(cath, out1T, out1b, nullptr, 1500, oh1, 1504, 1504, 2);           // selu
  gemm(oh1, out2T, out2b, out_o, 1200, oh2, 1200, 1504, 2);              // o (selu)
  gemm(oh2, WAT, WAb, out_feat, 1200, nullptr, 0, 1200, 0);              // feat
  gemm(xh, WccT, bcc, nullptr, 600, cc1h, 608, 320, 0);
  gemm(cc1h, Wcc2T, bcc2, qf, DD, nullptr, 0, 608, 0);                   // c_in -> qf
  k_reduce_lw<<<(2 * BB * DD + 255) / 256, 256, 0, stream>>>(x, Wm, bm, out_h);
  k_reduce_lw<<<(2 * BB * DD + 255) / 256, 256, 0, stream>>>(qf, Wn, bn, out_c);
}
